// DiffusionGraphGenerator_62809601736828
// MI455X (gfx1250) — matrix operand order repaired and checked by reading, not compiled, not run on hardware
//
#include <hip/hip_runtime.h>
#include <hip/hip_bf16.h>
#include <math.h>

// ---------------------------------------------------------------------------
// GraphDenoiser forward for MI455X (gfx1250, wave32, WMMA bf16).
// Hot path: per-edge message MLP (E=640k, 272->128->128) fused as
// gather -> WMMA GEMM -> SiLU -> WMMA GEMM -> atomic scatter-add, with
// packed weights staged in LDS and 5 edge-tiles per wave.
// ---------------------------------------------------------------------------

typedef __attribute__((ext_vector_type(16))) __bf16 v16bf;
typedef __attribute__((ext_vector_type(8)))  __bf16 v8bf;
typedef __attribute__((ext_vector_type(8)))  float  v8f;

#define NN 20000
#define EE 640000
#define LL 3
#define TPW 5   // edge tiles per wave in k_edge_msg

union AFrag { v16bf v; v8bf h[2]; };

__device__ __forceinline__ v8f wmma_bf16(v16bf a, v16bf b, v8f c) {
  // D = A(16x32) * B(32x16) + C, f32 accumulate
  return __builtin_amdgcn_wmma_f32_16x16x32_bf16(false, a, false, b,
                                                 (short)0, c, false, false);
}

// A fragment: row-major source row, lane holds row=lane&15,
// K pairs {kb..kb+7} and {kb+16..kb+23} -> two contiguous 16B loads.
__device__ __forceinline__ v16bf afrag_rows(const __bf16* rowbase, int kb) {
  AFrag f;
  f.h[0] = *(const v8bf*)(rowbase + kb);
  f.h[1] = *(const v8bf*)(rowbase + kb + 16);
  return f.v;
}

// B fragment from pre-packed weights (global or LDS): 32B contiguous per lane.
__device__ __forceinline__ v16bf bfrag(const __bf16* packed, int kt, int nt, int lane) {
  return ((const v16bf*)packed)[kt * 256 + nt * 32 + lane];
}

__device__ __forceinline__ float silu(float x) {
  return x / (1.0f + __expf(-x));
}

// ---------------------------------------------------------------------------
// Pack f32 weight W[Kin x 128] into B-fragment-native bf16 layout, K padded
// to Kp (multiple of 32). out[(kt*8+nt)*32*16 + lane*16 + j] =
//   W[(kt*32 + (j>>3)*16 + (lane>>4)*8 + (j&7)) * 128 + nt*16 + (lane&15)]
// ---------------------------------------------------------------------------
__global__ void k_pack_b(const float* __restrict__ W, int Kin, int Kp,
                         __bf16* __restrict__ out) {
  int t = blockIdx.x * blockDim.x + threadIdx.x;
  if (t >= Kp * 128) return;
  int j    = t & 15;
  int lane = (t >> 4) & 31;
  int nt   = (t >> 9) & 7;
  int kt   = t >> 12;
  int k = kt * 32 + ((j >> 3) << 4) + ((lane >> 4) << 3) + (j & 7);
  int n = nt * 16 + (lane & 15);
  out[t] = (k < Kin) ? (__bf16)W[k * 128 + n] : (__bf16)0.0f;
}

// ---------------------------------------------------------------------------
// Time embedding + 2-layer MLP (tiny): te[128]
// ---------------------------------------------------------------------------
__global__ void k_time(const int* __restrict__ tptr,
                       const float* __restrict__ w1, const float* __restrict__ b1,
                       const float* __restrict__ w2, const float* __restrict__ b2,
                       float* __restrict__ te) {
  __shared__ float emb[128];
  __shared__ float hid[128];
  int j = threadIdx.x;
  float t = (float)tptr[0];
  float lg = logf(10000.0f);
  if (j < 64) {
    float f = __expf(-lg * (float)j / 63.0f);
    emb[j] = __sinf(t * f);
  } else {
    float f = __expf(-lg * (float)(j - 64) / 63.0f);
    emb[j] = __cosf(t * f);
  }
  __syncthreads();
  float s = b1[j];
  for (int k = 0; k < 128; ++k) s += emb[k] * w1[k * 128 + j];
  hid[j] = silu(s);
  __syncthreads();
  float s2 = b2[j];
  for (int k = 0; k < 128; ++k) s2 += hid[k] * w2[k * 128 + j];
  te[j] = s2;
}

// ---------------------------------------------------------------------------
// Edge geometry: rbf[E x 16] in bf16 (fixed across layers)
// ---------------------------------------------------------------------------
__global__ void k_rbf(const float* __restrict__ x,
                      const int* __restrict__ srcI, const int* __restrict__ dstI,
                      __bf16* __restrict__ rbf) {
  int e = blockIdx.x * blockDim.x + threadIdx.x;
  if (e >= EE) return;
  int s = srcI[e], d = dstI[e];
  float dx = x[s * 3 + 0] - x[d * 3 + 0];
  float dy = x[s * 3 + 1] - x[d * 3 + 1];
  float r = sqrtf(dx * dx + dy * dy + 1e-8f);
  const float step  = 1.41421356237309515f / 15.0f;  // R_MAX/(R-1)
  const float gamma = 56.25f;                        // 1/(2*step^2)
  #pragma unroll
  for (int i = 0; i < 16; ++i) {
    float diff = r - step * (float)i;
    rbf[(size_t)e * 16 + i] = (__bf16)__expf(-gamma * diff * diff);
  }
}

__global__ void k_zero(float* __restrict__ p, int n) {
  int i = blockIdx.x * blockDim.x + threadIdx.x;
  if (i < n) p[i] = 0.0f;
}

// ---------------------------------------------------------------------------
// Node-input MLP: h = mlp([x_t, te]) : (131 -> 128 -> 128), K padded to 160.
// 1 wave = 16 nodes. Writes h (f32) and hb (bf16).
// ---------------------------------------------------------------------------
__global__ void __launch_bounds__(256)
k_node_in(const float* __restrict__ x_t, const float* __restrict__ te,
          const __bf16* __restrict__ w1p, const float* __restrict__ b1,
          const __bf16* __restrict__ w2p, const float* __restrict__ b2,
          float* __restrict__ h, __bf16* __restrict__ hb, int n) {
  __shared__ __bf16 Atile[8][16 * 160];
  __shared__ __bf16 Act[8][16 * 128];
  const int w = threadIdx.x >> 5, lane = threadIdx.x & 31;
  const int tile = blockIdx.x * 8 + w;
  const int ntiles = n >> 4;
  const bool valid = tile < ntiles;
  const int t0 = valid ? tile : 0;
  const int hi = lane >> 4, colL = lane & 15;

  __bf16* at = &Atile[w][0];
  {  // each lane fills half of its row (row = colL, cols [hi*80, hi*80+80))
    int node = t0 * 16 + colL;
    for (int k = hi * 80; k < hi * 80 + 80; ++k) {
      float v = (k < 3) ? x_t[node * 3 + k] : ((k < 131) ? te[k - 3] : 0.0f);
      at[colL * 160 + k] = (__bf16)v;
    }
  }
  __syncthreads();

  v8f acc[8] = {};
  #pragma unroll
  for (int kt = 0; kt < 5; ++kt) {
    v16bf a = afrag_rows(at + colL * 160, kt * 32 + hi * 8);
    #pragma unroll
    for (int nt = 0; nt < 8; ++nt)
      acc[nt] = wmma_bf16(a, bfrag(w1p, kt, nt, lane), acc[nt]);
  }

  __bf16* act = &Act[w][0];
  #pragma unroll
  for (int nt = 0; nt < 8; ++nt) {
    int col = nt * 16 + colL;
    float bb = b1[col];
    #pragma unroll
    for (int r = 0; r < 8; ++r) {
      float x = acc[nt][r] + bb;
      act[(hi * 8 + r) * 128 + col] = (__bf16)silu(x);
    }
  }
  __syncthreads();

  v8f acc2[8] = {};
  #pragma unroll
  for (int kt = 0; kt < 4; ++kt) {
    v16bf a = afrag_rows(act + colL * 128, kt * 32 + hi * 8);
    #pragma unroll
    for (int nt = 0; nt < 8; ++nt)
      acc2[nt] = wmma_bf16(a, bfrag(w2p, kt, nt, lane), acc2[nt]);
  }
  #pragma unroll
  for (int nt = 0; nt < 8; ++nt) {
    int col = nt * 16 + colL;
    float bb = b2[col];
    #pragma unroll
    for (int r = 0; r < 8; ++r) {
      if (valid) {
        int node = tile * 16 + hi * 8 + r;
        float y = acc2[nt][r] + bb;
        h[(size_t)node * 128 + col] = y;
        hb[(size_t)node * 128 + col] = (__bf16)y;
      }
    }
  }
}

// ---------------------------------------------------------------------------
// THE HOT KERNEL: fused edge message MLP + scatter-add.
// Packed weights staged in LDS once per block; each wave processes TPW
// 16-edge tiles. Per tile: GEMM1 [16x288]x[288x128] (72 WMMAs), SiLU,
// per-wave LDS transpose (no barrier: intra-wave LDS is in-order),
// GEMM2 [16x128]x[128x128] (32 WMMAs), f32 atomic scatter into agg.
// LDS: 72K (W1) + 32K (W2) + 32K (Act) = 136 KB -> 2 blocks/WGP.
// ---------------------------------------------------------------------------
__global__ void __launch_bounds__(256)
k_edge_msg(const __bf16* __restrict__ hb, const __bf16* __restrict__ rbf,
           const int* __restrict__ srcI, const int* __restrict__ dstI,
           const __bf16* __restrict__ w1p, const float* __restrict__ b1,
           const __bf16* __restrict__ w2p, const float* __restrict__ b2,
           float* __restrict__ agg) {
  __shared__ __bf16 W1s[288 * 128];
  __shared__ __bf16 W2s[128 * 128];
  __shared__ __bf16 Act[8][16 * 128];
  const int tid = threadIdx.x;

  // cooperative weight staging (16B chunks)
  for (int i = tid; i < 288 * 128 / 8; i += 256)
    ((v8bf*)W1s)[i] = ((const v8bf*)w1p)[i];
  for (int i = tid; i < 128 * 128 / 8; i += 256)
    ((v8bf*)W2s)[i] = ((const v8bf*)w2p)[i];
  __syncthreads();

  const int w = tid >> 5, lane = tid & 31;
  const int hi = lane >> 4, colL = lane & 15;
  __bf16* act = &Act[w][0];

  float bias1[8], bias2[8];
  #pragma unroll
  for (int nt = 0; nt < 8; ++nt) {
    bias1[nt] = b1[nt * 16 + colL];
    bias2[nt] = b2[nt * 16 + colL];
  }

  const int waveGlobal = blockIdx.x * 8 + w;

  for (int t = 0; t < TPW; ++t) {
    const int tile = waveGlobal * TPW + t;
    const int ebase = tile * 16;

    const int ea = ebase + colL;  // A-fragment row owned by this lane
    const int sN = srcI[ea];
    const int dN = dstI[ea];
    const __bf16* srow = hb + (size_t)sN * 128;
    const __bf16* drow = hb + (size_t)dN * 128;
    const __bf16* rrow = rbf + (size_t)ea * 16;

    v8f acc[8] = {};
    #pragma unroll
    for (int kt = 0; kt < 9; ++kt) {
      v16bf a;
      if (kt < 4) {
        a = afrag_rows(srow, kt * 32 + hi * 8);
      } else if (kt < 8) {
        a = afrag_rows(drow, (kt - 4) * 32 + hi * 8);
      } else {  // rbf (16 vals) + 16 zero-pad
        AFrag f;
        f.h[0] = *(const v8bf*)(rrow + hi * 8);
        #pragma unroll
        for (int j = 0; j < 8; ++j) f.v[8 + j] = (__bf16)0.0f;
        a = f.v;
      }
      #pragma unroll
      for (int nt = 0; nt < 8; ++nt)
        acc[nt] = wmma_bf16(a, bfrag(W1s, kt, nt, lane), acc[nt]);
    }

    // bias + SiLU -> per-wave LDS tile (D layout -> row-major bf16)
    #pragma unroll
    for (int nt = 0; nt < 8; ++nt) {
      int col = nt * 16 + colL;
      #pragma unroll
      for (int r = 0; r < 8; ++r) {
        float x = acc[nt][r] + bias1[nt];
        act[(hi * 8 + r) * 128 + col] = (__bf16)silu(x);
      }
    }
    // no barrier needed: per-wave LDS ops execute in order

    v8f acc2[8] = {};
    #pragma unroll
    for (int kt = 0; kt < 4; ++kt) {
      v16bf a = afrag_rows(act + colL * 128, kt * 32 + hi * 8);
      #pragma unroll
      for (int nt = 0; nt < 8; ++nt)
        acc2[nt] = wmma_bf16(a, bfrag(W2s, kt, nt, lane), acc2[nt]);
    }

    // scatter-add: agg[dst[e], :] += m[e, :]
    int dn[8];
    #pragma unroll
    for (int r = 0; r < 8; ++r) dn[r] = dstI[ebase + hi * 8 + r];
    #pragma unroll
    for (int nt = 0; nt < 8; ++nt) {
      int col = nt * 16 + colL;
      #pragma unroll
      for (int r = 0; r < 8; ++r)
        atomicAdd(&agg[(size_t)dn[r] * 128 + col], acc2[nt][r] + bias2[nt]);
    }
  }
}

// ---------------------------------------------------------------------------
// Node update: h = LN(h + mlp([h, agg])) ; also refresh hb.
// 1 wave = 16 nodes. K = 256 (h bf16 | agg f32->bf16).
// ---------------------------------------------------------------------------
__global__ void __launch_bounds__(256)
k_phih(float* __restrict__ h, __bf16* __restrict__ hb,
       const float* __restrict__ agg,
       const __bf16* __restrict__ w1p, const float* __restrict__ b1,
       const __bf16* __restrict__ w2p, const float* __restrict__ b2,
       const float* __restrict__ g, const float* __restrict__ bln, int n) {
  __shared__ __bf16 Act[8][16 * 128];
  __shared__ float  Ln[8][16 * 128];
  const int w = threadIdx.x >> 5, lane = threadIdx.x & 31;
  const int tile = blockIdx.x * 8 + w;
  const int ntiles = n >> 4;
  const bool valid = tile < ntiles;
  const int t0 = valid ? tile : 0;
  const int hi = lane >> 4, colL = lane & 15;

  const int nodeA = t0 * 16 + colL;
  const __bf16* hrow = hb + (size_t)nodeA * 128;
  const float*  arow = agg + (size_t)nodeA * 128;

  v8f acc[8] = {};
  #pragma unroll
  for (int kt = 0; kt < 8; ++kt) {
    v16bf a;
    if (kt < 4) {
      a = afrag_rows(hrow, kt * 32 + hi * 8);
    } else {  // agg: f32 -> bf16 on the fly
      AFrag f;
      int kb = (kt - 4) * 32 + hi * 8;
      #pragma unroll
      for (int j = 0; j < 8; ++j) f.v[j]     = (__bf16)arow[kb + j];
      #pragma unroll
      for (int j = 0; j < 8; ++j) f.v[8 + j] = (__bf16)arow[kb + 16 + j];
      a = f.v;
    }
    #pragma unroll
    for (int nt = 0; nt < 8; ++nt)
      acc[nt] = wmma_bf16(a, bfrag(w1p, kt, nt, lane), acc[nt]);
  }

  __bf16* act = &Act[w][0];
  #pragma unroll
  for (int nt = 0; nt < 8; ++nt) {
    int col = nt * 16 + colL;
    float bb = b1[col];
    #pragma unroll
    for (int r = 0; r < 8; ++r) {
      float x = acc[nt][r] + bb;
      act[(hi * 8 + r) * 128 + col] = (__bf16)silu(x);
    }
  }
  __syncthreads();

  v8f acc2[8] = {};
  #pragma unroll
  for (int kt = 0; kt < 4; ++kt) {
    v16bf a = afrag_rows(act + colL * 128, kt * 32 + hi * 8);
    #pragma unroll
    for (int nt = 0; nt < 8; ++nt)
      acc2[nt] = wmma_bf16(a, bfrag(w2p, kt, nt, lane), acc2[nt]);
  }

  // residual into LDS tile for LayerNorm
  float* ln = &Ln[w][0];
  #pragma unroll
  for (int nt = 0; nt < 8; ++nt) {
    int col = nt * 16 + colL;
    float bb = b2[col];
    #pragma unroll
    for (int r = 0; r < 8; ++r) {
      int row = hi * 8 + r;
      int node = t0 * 16 + row;
      ln[row * 128 + col] = acc2[nt][r] + bb + h[(size_t)node * 128 + col];
    }
  }
  __syncthreads();

  if (lane < 16 && valid) {
    const int row = lane;
    const int node = tile * 16 + row;
    const float* x = ln + row * 128;
    float mu = 0.0f, m2 = 0.0f;
    #pragma unroll 4
    for (int k = 0; k < 128; ++k) { float v = x[k]; mu += v; m2 += v * v; }
    mu *= (1.0f / 128.0f);
    m2 *= (1.0f / 128.0f);
    float rstd = rsqrtf(m2 - mu * mu + 1e-5f);
    for (int k = 0; k < 128; ++k) {
      float y = (x[k] - mu) * rstd * g[k] + bln[k];
      h[(size_t)node * 128 + k] = y;
      hb[(size_t)node * 128 + k] = (__bf16)y;
    }
  }
}

// ---------------------------------------------------------------------------
// Output head: eps = mlp(h) : 128 -> 128 -> 3 (small; plain VALU)
// ---------------------------------------------------------------------------
__global__ void k_out(const float* __restrict__ h,
                      const float* __restrict__ w1, const float* __restrict__ b1,
                      const float* __restrict__ w2, const float* __restrict__ b2,
                      float* __restrict__ out) {
  __shared__ float row[128];
  __shared__ float hid[128];
  int node = blockIdx.x;
  int j = threadIdx.x;
  row[j] = h[(size_t)node * 128 + j];
  __syncthreads();
  float s = b1[j];
  for (int k = 0; k < 128; ++k) s += row[k] * w1[k * 128 + j];
  hid[j] = silu(s);
  __syncthreads();
  if (j < 3) {
    float s2 = b2[j];
    for (int k = 0; k < 128; ++k) s2 += hid[k] * w2[k * 3 + j];
    out[(size_t)node * 3 + j] = s2;
  }
}

// ---------------------------------------------------------------------------
extern "C" void kernel_launch(void* const* d_in, const int* in_sizes, int n_in,
                              void* d_out, int out_size, void* d_ws, size_t ws_size,
                              hipStream_t stream) {
  (void)in_sizes; (void)n_in; (void)out_size; (void)ws_size;
  const float* x_t   = (const float*)d_in[0];
  const int*   eidx  = (const int*)d_in[1];
  const int*   tptr  = (const int*)d_in[2];
  const float* tw1   = (const float*)d_in[3];
  const float* tb1   = (const float*)d_in[4];
  const float* tw2   = (const float*)d_in[5];
  const float* tb2   = (const float*)d_in[6];
  const float* niw1  = (const float*)d_in[7];
  const float* nib1  = (const float*)d_in[8];
  const float* niw2  = (const float*)d_in[9];
  const float* nib2  = (const float*)d_in[10];
  const float* pmw1  = (const float*)d_in[11];
  const float* pmb1  = (const float*)d_in[12];
  const float* pmw2  = (const float*)d_in[13];
  const float* pmb2  = (const float*)d_in[14];
  const float* phw1  = (const float*)d_in[15];
  const float* phb1  = (const float*)d_in[16];
  const float* phw2  = (const float*)d_in[17];
  const float* phb2  = (const float*)d_in[18];
  const float* lng   = (const float*)d_in[19];
  const float* lnb   = (const float*)d_in[20];
  const float* ow1   = (const float*)d_in[21];
  const float* ob1   = (const float*)d_in[22];
  const float* ow2   = (const float*)d_in[23];
  const float* ob2   = (const float*)d_in[24];

  const int* srcI = eidx;
  const int* dstI = eidx + EE;

  // ---- workspace carve-up (256B aligned) ----
  char* base = (char*)d_ws;
  size_t off = 0;
  auto carve = [&](size_t bytes) {
    void* p = base + off;
    off = (off + bytes + 255) & ~(size_t)255;
    return p;
  };
  float*  te    = (float*)carve(128 * 4);
  float*  h     = (float*)carve((size_t)NN * 128 * 4);
  __bf16* hbb   = (__bf16*)carve((size_t)NN * 128 * 2);
  float*  agg   = (float*)carve((size_t)NN * 128 * 4);
  __bf16* rbf   = (__bf16*)carve((size_t)EE * 16 * 2);
  __bf16* niw1p = (__bf16*)carve((size_t)160 * 128 * 2);
  __bf16* niw2p = (__bf16*)carve((size_t)128 * 128 * 2);
  __bf16* pmw1p = (__bf16*)carve((size_t)LL * 288 * 128 * 2);
  __bf16* pmw2p = (__bf16*)carve((size_t)LL * 128 * 128 * 2);
  __bf16* phw1p = (__bf16*)carve((size_t)LL * 256 * 128 * 2);
  __bf16* phw2p = (__bf16*)carve((size_t)LL * 128 * 128 * 2);

  auto packGrid = [](int Kp) { return dim3((Kp * 128 + 255) / 256); };

  // ---- pack weights into B-fragment bf16 layout ----
  k_pack_b<<<packGrid(160), 256, 0, stream>>>(niw1, 131, 160, niw1p);
  k_pack_b<<<packGrid(128), 256, 0, stream>>>(niw2, 128, 128, niw2p);
  for (int l = 0; l < LL; ++l) {
    k_pack_b<<<packGrid(288), 256, 0, stream>>>(pmw1 + (size_t)l * 272 * 128, 272, 288,
                                                pmw1p + (size_t)l * 288 * 128);
    k_pack_b<<<packGrid(128), 256, 0, stream>>>(pmw2 + (size_t)l * 128 * 128, 128, 128,
                                                pmw2p + (size_t)l * 128 * 128);
    k_pack_b<<<packGrid(256), 256, 0, stream>>>(phw1 + (size_t)l * 256 * 128, 256, 256,
                                                phw1p + (size_t)l * 256 * 128);
    k_pack_b<<<packGrid(128), 256, 0, stream>>>(phw2 + (size_t)l * 128 * 128, 128, 128,
                                                phw2p + (size_t)l * 128 * 128);
  }

  // ---- time embedding MLP, edge RBFs ----
  k_time<<<1, 128, 0, stream>>>(tptr, tw1, tb1, tw2, tb2, te);
  k_rbf<<<(EE + 255) / 256, 256, 0, stream>>>(x_t, srcI, dstI, rbf);

  // ---- node input MLP ----
  const int nodeTiles = NN / 16;                       // 1250
  const int nodeBlocks = (nodeTiles + 7) / 8;          // 157
  k_node_in<<<nodeBlocks, 256, 0, stream>>>(x_t, te, niw1p, nib1, niw2p, nib2,
                                            h, hbb, NN);

  // ---- message-passing layers ----
  const int edgeBlocks = EE / 16 / 8 / TPW;            // 1000 (exact)
  for (int l = 0; l < LL; ++l) {
    k_zero<<<(NN * 128 + 255) / 256, 256, 0, stream>>>(agg, NN * 128);
    k_edge_msg<<<edgeBlocks, 256, 0, stream>>>(
        hbb, rbf, srcI, dstI,
        pmw1p + (size_t)l * 288 * 128, pmb1 + (size_t)l * 128,
        pmw2p + (size_t)l * 128 * 128, pmb2 + (size_t)l * 128, agg);
    k_phih<<<nodeBlocks, 256, 0, stream>>>(
        h, hbb, agg,
        phw1p + (size_t)l * 256 * 128, phb1 + (size_t)l * 128,
        phw2p + (size_t)l * 128 * 128, phb2 + (size_t)l * 128,
        lng + (size_t)l * 128, lnb + (size_t)l * 128, NN);
  }

  // ---- output head ----
  k_out<<<NN, 128, 0, stream>>>(h, ow1, ob1, ow2, ob2, (float*)d_out);
}
